// Comb2_79379585565128
// MI455X (gfx1250) — matrix operand order repaired and checked by reading, not compiled, not run on hardware
//
#include <hip/hip_runtime.h>

// ---------------------------------------------------------------------------
// CDNA5 (gfx1250) implementation. All H x H linears lower to
// v_wmma_f32_16x16x32_bf16 (f32 accumulate). wave32 throughout.
// Weights are pre-packed to bf16 in WMMA B-fragment order so every B
// fragment is a single aligned 32-byte vector load. Output stores use a
// per-wave full-tile fast path (no per-element exec-mask branching).
// ---------------------------------------------------------------------------

typedef __attribute__((ext_vector_type(16))) __bf16 v16bf;
typedef __attribute__((ext_vector_type(8)))  float  v8f;

static inline int cdiv(int a, int b) { return (a + b - 1) / b; }

// Order-preserving float -> u32 key so segment-max can be one u32 atomicMax.
__device__ __forceinline__ unsigned fkey(float f) {
  unsigned u = __float_as_uint(f);
  return (u & 0x80000000u) ? ~u : (u | 0x80000000u);
}
__device__ __forceinline__ float funkey(unsigned u) {
  return (u & 0x80000000u) ? __uint_as_float(u & 0x7FFFFFFFu) : __uint_as_float(~u);
}

// --------------------------- utility: zero fill ----------------------------
__global__ void zero_kernel(unsigned* __restrict__ p, size_t n) {
  size_t i = (size_t)blockIdx.x * blockDim.x + threadIdx.x;
  size_t stride = (size_t)gridDim.x * blockDim.x;
  for (; i < n; i += stride) p[i] = 0u;
}

// ---------------- weight pre-pack: f32 (K x 64) -> bf16 fragments ----------
// Packed layout: off = ((ct*(K/32) + kb)*32 + lane)*16 + i, where
//   k = kb*32 + (i>=8 ? 16 : 0) + (lane>=16 ? 8 : 0) + (i&7),  col = ct*16 + (lane&15)
__global__ void pack_w_kernel(const float* __restrict__ W, int K,
                              __bf16* __restrict__ out, int ctOff) {
  int t = blockIdx.x * blockDim.x + threadIdx.x;
  if (t >= K * 64) return;
  int k = t / 64, col = t % 64;
  int kb = k >> 5;
  bool hi = (k & 8) != 0;
  int i = (k & 7) + ((k & 16) ? 8 : 0);
  int lane = (col & 15) + (hi ? 16 : 0);
  int ct = (col >> 4) + ctOff;
  int nkb = K >> 5;
  size_t off = (((size_t)ct * nkb + kb) * 32 + lane) * 16 + i;
  out[off] = (__bf16)W[t];
}

// ----------------- segment statistics (sum, sumsq, count) ------------------
template <int F>
__global__ void seg_stats_kernel(const float* __restrict__ x,
                                 const int* __restrict__ batch,
                                 const int* __restrict__ idx, int n,
                                 float* __restrict__ gsum, float* __restrict__ gsq,
                                 float* __restrict__ gcnt) {
  __shared__ float ssum[64 * F];
  __shared__ float ssq[64 * F];
  __shared__ float scnt[64];
  for (int t = threadIdx.x; t < 64 * F; t += blockDim.x) { ssum[t] = 0.f; ssq[t] = 0.f; }
  for (int t = threadIdx.x; t < 64; t += blockDim.x) scnt[t] = 0.f;
  __syncthreads();
  int i = blockIdx.x * blockDim.x + threadIdx.x;
  if (i < n) {
    int g = batch[idx ? idx[i] : i];
    atomicAdd(&scnt[g], 1.0f);
#pragma unroll
    for (int f = 0; f < F; f++) {
      float v = x[(size_t)i * F + f];
      atomicAdd(&ssum[g * F + f], v);
      atomicAdd(&ssq[g * F + f], v * v);
    }
  }
  __syncthreads();
  for (int t = threadIdx.x; t < 64 * F; t += blockDim.x) {
    if (ssum[t] != 0.f) atomicAdd(&gsum[t], ssum[t]);
    if (ssq[t] != 0.f) atomicAdd(&gsq[t], ssq[t]);
  }
  for (int t = threadIdx.x; t < 64; t += blockDim.x)
    if (scnt[t] != 0.f) atomicAdd(&gcnt[t], scnt[t]);
}

// graph_norm / instance_norm apply.
template <int F>
__global__ void norm_apply_kernel(const float* __restrict__ x,
                                  const int* __restrict__ batch,
                                  const int* __restrict__ idx, int n,
                                  const float* __restrict__ gsum,
                                  const float* __restrict__ gsq,
                                  const float* __restrict__ gcnt,
                                  const float* __restrict__ w,
                                  const float* __restrict__ b,
                                  const float* __restrict__ ms,
                                  float* __restrict__ out) {
  int t = blockIdx.x * blockDim.x + threadIdx.x;
  if (t >= n * F) return;
  int i = t / F, f = t % F;
  int g = batch[idx ? idx[i] : i];
  float c = fmaxf(gcnt[g], 1.0f);
  float m = gsum[g * F + f] / c;
  float msq = gsq[g * F + f] / c;
  float s = ms ? ms[f] : 1.0f;
  float var = msq - 2.f * s * m * m + s * s * m * m;
  float xc = x[t] - s * m;
  float r = xc * rsqrtf(var + 1e-5f);
  out[t] = (w ? w[f] : 1.f) * r + (b ? b[f] : 0.f);
}

// ------------------ small-K input linear (K<=12) + ReLU --------------------
__global__ void lin_small_kernel(const float* __restrict__ x, int n, int fin,
                                 const float* __restrict__ W,
                                 const float* __restrict__ b,
                                 float* __restrict__ y) {
  int t = blockIdx.x * blockDim.x + threadIdx.x;
  if (t >= n * 64) return;
  int i = t >> 6, o = t & 63;
  float acc = b[o];
  for (int f = 0; f < fin; f++) acc += x[(size_t)i * fin + f] * W[f * 64 + o];
  y[t] = fmaxf(acc, 0.f);
}

// Load one A fragment (16-bit A 16x32 layout, ISA 7.12.2) from a f32 row.
__device__ __forceinline__ v16bf load_a_frag(const float* __restrict__ arow,
                                             int kb, bool hi) {
  int ka = kb * 32 + (hi ? 8 : 0);
  float4 f0 = *(const float4*)(arow + ka);
  float4 f1 = *(const float4*)(arow + ka + 4);
  float4 f2 = *(const float4*)(arow + ka + 16);
  float4 f3 = *(const float4*)(arow + ka + 20);
  v16bf a;
  a[0] = (__bf16)f0.x;  a[1] = (__bf16)f0.y;  a[2] = (__bf16)f0.z;  a[3] = (__bf16)f0.w;
  a[4] = (__bf16)f1.x;  a[5] = (__bf16)f1.y;  a[6] = (__bf16)f1.z;  a[7] = (__bf16)f1.w;
  a[8] = (__bf16)f2.x;  a[9] = (__bf16)f2.y;  a[10] = (__bf16)f2.z; a[11] = (__bf16)f2.w;
  a[12] = (__bf16)f3.x; a[13] = (__bf16)f3.y; a[14] = (__bf16)f3.z; a[15] = (__bf16)f3.w;
  return a;
}

// Store one 16x16 f32 D tile (v8f per lane) with full-tile fast path.
template <int OSTRIDE, bool RELU>
__device__ __forceinline__ void store_d_tile(float* __restrict__ Y, int rowBase,
                                             int col, bool hi, bool full, int n,
                                             v8f c, float bv) {
  if (full) {
    float* yp = Y + (size_t)(rowBase + (hi ? 8 : 0)) * OSTRIDE + col;
#pragma unroll
    for (int r = 0; r < 8; r++) {
      float vv = c[r] + bv;
      if (RELU) vv = fmaxf(vv, 0.f);
      yp[(size_t)r * OSTRIDE] = vv;
    }
  } else {
#pragma unroll
    for (int r = 0; r < 8; r++) {
      int orow = rowBase + r + (hi ? 8 : 0);
      if (orow < n) {
        float vv = c[r] + bv;
        if (RELU) vv = fmaxf(vv, 0.f);
        Y[(size_t)orow * OSTRIDE + col] = vv;
      }
    }
  }
}

// -------- fused QKVS WMMA GEMM: Y[n][256] = A[n][64] @ [Wq|Wk|Wv|Ws] -------
__global__ void __launch_bounds__(256)
gemm_qkvs_wmma_kernel(const float* __restrict__ A, const __bf16* __restrict__ Wp,
                      const float* __restrict__ b0, const float* __restrict__ b1,
                      const float* __restrict__ b2, const float* __restrict__ b3,
                      float* __restrict__ Y, int n) {
  int lane = threadIdx.x & 31;
  int wave = threadIdx.x >> 5;
  int rowTile = blockIdx.x * 8 + wave;
  bool hi = lane >= 16;
  int mloc = lane & 15;
  int rowBase = rowTile * 16;
  bool full = rowBase + 16 <= n;
  int row = rowBase + mloc;
  int ar = row < n ? row : (n - 1);
  const float* arow = A + (size_t)ar * 64;
  __builtin_prefetch(arow + (size_t)128 * 64, 0, 1);  // next block's rows
  v16bf a0 = load_a_frag(arow, 0, hi);
  v16bf a1 = load_a_frag(arow, 1, hi);
#pragma unroll
  for (int ct = 0; ct < 16; ct++) {
    v8f c = {};
    v16bf w0 = *(const v16bf*)(Wp + (((size_t)ct * 2 + 0) * 32 + lane) * 16);
    v16bf w1 = *(const v16bf*)(Wp + (((size_t)ct * 2 + 1) * 32 + lane) * 16);
    c = __builtin_amdgcn_wmma_f32_16x16x32_bf16(false, a0, false, w0, (short)0, c, false, false);
    c = __builtin_amdgcn_wmma_f32_16x16x32_bf16(false, a1, false, w1, (short)0, c, false, false);
    const float* bp = (ct < 4) ? b0 : (ct < 8) ? b1 : (ct < 12) ? b2 : b3;
    float bv = bp[(ct & 3) * 16 + mloc];
    store_d_tile<256, false>(Y, rowBase, ct * 16 + mloc, hi, full, n, c, bv);
  }
}

// -------- single-output WMMA GEMM: Y[n][64] = A[gather][64] @ W + b --------
template <bool GATHER>
__global__ void __launch_bounds__(256)
gemm64_wmma_kernel(const float* __restrict__ A, const int* __restrict__ idx,
                   const __bf16* __restrict__ Wp, const float* __restrict__ bias,
                   float* __restrict__ Y, int n) {
  int lane = threadIdx.x & 31;
  int wave = threadIdx.x >> 5;
  int rowTile = blockIdx.x * 8 + wave;
  bool hi = lane >= 16;
  int mloc = lane & 15;
  int rowBase = rowTile * 16;
  bool full = rowBase + 16 <= n;
  int row = rowBase + mloc;
  int ar = row < n ? row : (n - 1);
  if (GATHER) ar = idx[ar];
  const float* arow = A + (size_t)ar * 64;
  __builtin_prefetch(arow + (size_t)128 * 64, 0, 1);
  v16bf a0 = load_a_frag(arow, 0, hi);
  v16bf a1 = load_a_frag(arow, 1, hi);
#pragma unroll
  for (int ct = 0; ct < 4; ct++) {
    v8f c = {};
    v16bf w0 = *(const v16bf*)(Wp + (((size_t)ct * 2 + 0) * 32 + lane) * 16);
    v16bf w1 = *(const v16bf*)(Wp + (((size_t)ct * 2 + 1) * 32 + lane) * 16);
    c = __builtin_amdgcn_wmma_f32_16x16x32_bf16(false, a0, false, w0, (short)0, c, false, false);
    c = __builtin_amdgcn_wmma_f32_16x16x32_bf16(false, a1, false, w1, (short)0, c, false, false);
    int col = ct * 16 + mloc;
    store_d_tile<64, false>(Y, rowBase, col, hi, full, n, c, bias[col]);
  }
}

// ------- WMMA edge MLP: Zout = relu(concat(x[row],x[col],z) @ W(192x64)+b) -
__global__ void __launch_bounds__(256)
edge_mlp_wmma_kernel(const float* __restrict__ X, const float* __restrict__ Z,
                     const int* __restrict__ erow, const int* __restrict__ ecol,
                     const __bf16* __restrict__ Wp, const float* __restrict__ bias,
                     float* __restrict__ Zout, int E) {
  int lane = threadIdx.x & 31;
  int wave = threadIdx.x >> 5;
  int eTile = blockIdx.x * 8 + wave;
  bool hi = lane >= 16;
  int mloc = lane & 15;
  int eBase = eTile * 16;
  bool full = eBase + 16 <= E;
  int eloc = eBase + mloc;
  int e = eloc < E ? eloc : (E - 1);
  int r0 = erow[e], c0 = ecol[e];
  const float* xr = X + (size_t)r0 * 64;
  const float* xc = X + (size_t)c0 * 64;
  const float* zr = Z + (size_t)e * 64;

  v16bf a[6];
#pragma unroll
  for (int kb = 0; kb < 6; kb++) {
#pragma unroll
    for (int i = 0; i < 16; i++) {
      int kk = kb * 32 + ((i < 8) ? 0 : 16) + (hi ? 8 : 0) + (i & 7);
      float f = (kk < 64) ? xr[kk] : ((kk < 128) ? xc[kk - 64] : zr[kk - 128]);
      a[kb][i] = (__bf16)f;
    }
  }
#pragma unroll
  for (int ct = 0; ct < 4; ct++) {
    v8f c = {};
#pragma unroll
    for (int kb = 0; kb < 6; kb++) {
      v16bf w = *(const v16bf*)(Wp + (((size_t)ct * 6 + kb) * 32 + lane) * 16);
      c = __builtin_amdgcn_wmma_f32_16x16x32_bf16(false, a[kb], false, w, (short)0, c, false, false);
    }
    int col = ct * 16 + mloc;
    store_d_tile<64, true>(Zout, eBase, col, hi, full, E, c, bias[col]);
  }
}

// -------------------- attention: per-edge logit + seg max ------------------
// qkvs layout: [n][256] = [q(64) | k(64) | v(64) | s(64)]
__global__ void attn_logits_kernel(const int* __restrict__ erow,
                                   const int* __restrict__ ecol,
                                   const float* __restrict__ qkvs,
                                   const float* __restrict__ ep,
                                   float* __restrict__ logits,
                                   unsigned* __restrict__ umax, int E) {
  int lane = threadIdx.x & 31;
  int e = blockIdx.x * 8 + (threadIdx.x >> 5);
  if (e >= E) return;
  int r = erow[e], ci = ecol[e];
  float k0 = qkvs[(size_t)r * 256 + 64 + lane];
  float k1 = qkvs[(size_t)r * 256 + 96 + lane];
  if (ep) {
    k0 += ep[(size_t)e * 64 + lane];
    k1 += ep[(size_t)e * 64 + lane + 32];
  }
  float acc = qkvs[(size_t)ci * 256 + lane] * k0 +
              qkvs[(size_t)ci * 256 + 32 + lane] * k1;
#pragma unroll
  for (int s = 16; s > 0; s >>= 1) acc += __shfl_xor(acc, s, 32);
  if (lane == 0) {
    float lg = acc * 0.125f;  // 1/sqrt(64)
    logits[e] = lg;
    atomicMax(&umax[ci], fkey(lg));
  }
}

// ---- per-edge softmax numerator: logits[e] <- exp(logit-max), sums += -----
__global__ void attn_weight_kernel(const int* __restrict__ ecol,
                                   float* __restrict__ logits,
                                   const unsigned* __restrict__ umax,
                                   float* __restrict__ sums, int E) {
  int e = blockIdx.x * blockDim.x + threadIdx.x;
  if (e >= E) return;
  int ci = ecol[e];
  float w = expf(logits[e] - funkey(umax[ci]));
  logits[e] = w;
  atomicAdd(&sums[ci], w);
}

// --------- attention: scatter unnormalized w*v into target nodes -----------
__global__ void attn_scatter_kernel(const int* __restrict__ erow,
                                    const int* __restrict__ ecol,
                                    const float* __restrict__ qkvs,
                                    const float* __restrict__ ep,
                                    const float* __restrict__ wts,
                                    float* __restrict__ aggr, int E) {
  long long t = (long long)blockIdx.x * blockDim.x + threadIdx.x;
  if (t >= (long long)E * 64) return;
  int e = (int)(t >> 6), ch = (int)(t & 63);
  int r = erow[e], ci = ecol[e];
  float w = wts[e];
  float ve = qkvs[(size_t)r * 256 + 128 + ch];
  if (ep) ve += ep[(size_t)e * 64 + ch];
  atomicAdd(&aggr[(size_t)ci * 64 + ch], w * ve);
}

// ------------- node finalize: divide, add skip, relu or mask ---------------
template <bool MASKED>
__global__ void finalize_kernel(const float* __restrict__ aggr,
                                const float* __restrict__ sums,
                                const float* __restrict__ qkvs,
                                const unsigned char* __restrict__ mask,
                                float* __restrict__ y, int n) {
  long long t = (long long)blockIdx.x * blockDim.x + threadIdx.x;
  if (t >= (long long)n * 64) return;
  int i = (int)(t >> 6), ch = (int)(t & 63);
  float val = aggr[t] / (sums[i] + 1e-16f) + qkvs[(size_t)i * 256 + 192 + ch];
  if (MASKED) {
    if (mask[i]) y[t] = val;  // else keep old value
  } else {
    y[t] = fmaxf(val, 0.f);
  }
}

// ------------------------- tail: (P x 128) @ (128 x 2) ---------------------
__global__ void tail_kernel(const float* __restrict__ xcq,
                            const float* __restrict__ xdag,
                            const float* __restrict__ W,
                            const float* __restrict__ b,
                            float* __restrict__ out, int P) {
  int p = blockIdx.x * blockDim.x + threadIdx.x;
  if (p >= P) return;
  float o0 = b[0], o1 = b[1];
  for (int h = 0; h < 64; h++) {
    float v = fmaxf(xcq[(size_t)p * 64 + h], 0.f);
    o0 += v * W[h * 2 + 0];
    o1 += v * W[h * 2 + 1];
  }
  for (int h = 0; h < 64; h++) {
    float v = fmaxf(xdag[(size_t)p * 64 + h], 0.f);
    o0 += v * W[(64 + h) * 2 + 0];
    o1 += v * W[(64 + h) * 2 + 1];
  }
  out[p * 2 + 0] = o0;
  out[p * 2 + 1] = o1;
}

// ---------------------------------------------------------------------------
struct Dense { const float* W; const float* b; };

extern "C" void kernel_launch(void* const* d_in, const int* in_sizes, int n_in,
                              void* d_out, int out_size, void* d_ws, size_t ws_size,
                              hipStream_t stream) {
  const int G = 64;
  const int N1 = in_sizes[0] / 10;
  const int E1 = in_sizes[1] / 2;
  const int P  = in_sizes[6];
  const int N2 = in_sizes[7] / 6;
  const int E2 = in_sizes[8] / 2;

  const float* dag_x   = (const float*)d_in[0];
  const int*   dag_eu  = (const int*)d_in[1];
  const int*   dag_ed  = (const int*)d_in[2];
  const unsigned char* mask_up = (const unsigned char*)d_in[3];
  const unsigned char* mask_dn = (const unsigned char*)d_in[4];
  const int*   dag_batch = (const int*)d_in[5];
  const int*   dag_pick  = (const int*)d_in[6];
  const float* cq_x    = (const float*)d_in[7];
  const int*   cq_ei   = (const int*)d_in[8];
  const float* cq_ea   = (const float*)d_in[9];
  const int*   cq_batch = (const int*)d_in[10];
  const int*   cq_pick  = (const int*)d_in[11];
  const int* erow = cq_ei;       // source
  const int* ecol = cq_ei + E2;  // target

  // ---- param leaves in jax pytree (alphabetical) order, starting at 12 ----
  int pi = 12;
  auto D = [&]() { Dense d; d.W = (const float*)d_in[pi++]; d.b = (const float*)d_in[pi++]; return d; };
  Dense cq_e[3], cq_k[3], cq_q[3], cq_s[3], cq_v[3];
  for (int l = 0; l < 3; l++) { cq_e[l] = D(); cq_k[l] = D(); cq_q[l] = D(); cq_s[l] = D(); cq_v[l] = D(); }
  Dense emlp[2]; emlp[0] = D(); emlp[1] = D();
  const float* gne_b = (const float*)d_in[pi++];
  const float* gne_ms = (const float*)d_in[pi++];
  const float* gne_w = (const float*)d_in[pi++];
  const float* gnx_b = (const float*)d_in[pi++];
  const float* gnx_ms = (const float*)d_in[pi++];
  const float* gnx_w = (const float*)d_in[pi++];
  Dense cq_lin_edge = D(), cq_lin_in = D(), cq_lin_out = D();
  Dense dg_k[6], dg_q[6], dg_s[6], dg_v[6];
  for (int l = 0; l < 6; l++) { dg_k[l] = D(); dg_q[l] = D(); dg_s[l] = D(); dg_v[l] = D(); }
  Dense dg_lin_in = D(), dg_lin_out = D();
  Dense top = D();

  // ----------------------- workspace bump allocator ------------------------
  char* wp = (char*)d_ws;
  auto alloc = [&](size_t bytes) -> void* {
    void* p = (void*)wp;
    wp += (bytes + 255) & ~(size_t)255;
    return p;
  };
  const int N = (N1 > N2 ? N1 : N2);
  float* xb   = (float*)alloc((size_t)N * 64 * 4);    // node features
  float* qkvs = (float*)alloc((size_t)N * 256 * 4);   // fused q|k|v|s (also norm staging)
  float* aggr = (float*)alloc((size_t)N * 64 * 4);
  float* zb0  = (float*)alloc((size_t)E2 * 64 * 4);   // edge feat double buffer
  float* zb1  = (float*)alloc((size_t)E2 * 64 * 4);   // "next" buffer also hosts ep
  float* logits = (float*)alloc((size_t)E2 * 4);
  float* sums   = (float*)alloc((size_t)N * 4);
  unsigned* umax = (unsigned*)alloc((size_t)N * 4);
  float* xcq  = (float*)alloc((size_t)P * 64 * 4);
  float* xdag = (float*)alloc((size_t)P * 64 * 4);
  float* stats = (float*)alloc((size_t)(G * 12 * 2 + G) * 4);
  float* st_sum = stats;
  float* st_sq  = stats + G * 12;
  float* st_cnt = stats + 2 * G * 12;
  // packed bf16 weights
  __bf16* pk_cqkvs[3]; for (int l = 0; l < 3; l++) pk_cqkvs[l] = (__bf16*)alloc(16384 * 2);
  __bf16* pk_ce[3];    for (int l = 0; l < 3; l++) pk_ce[l]    = (__bf16*)alloc(4096 * 2);
  __bf16* pk_mlp[2];   for (int l = 0; l < 2; l++) pk_mlp[l]   = (__bf16*)alloc(12288 * 2);
  __bf16* pk_cout = (__bf16*)alloc(4096 * 2);
  __bf16* pk_dqkvs[6]; for (int l = 0; l < 6; l++) pk_dqkvs[l] = (__bf16*)alloc(16384 * 2);
  __bf16* pk_dout = (__bf16*)alloc(4096 * 2);

  auto zero = [&](void* p, size_t nwords) {
    size_t blk = (nwords + 255) / 256;
    int blocks = (int)(blk < 4096 ? (blk ? blk : 1) : 4096);
    zero_kernel<<<blocks, 256, 0, stream>>>((unsigned*)p, nwords);
  };
  auto pack = [&](const float* W, int K, __bf16* out, int ctOff) {
    pack_w_kernel<<<cdiv(K * 64, 256), 256, 0, stream>>>(W, K, out, ctOff);
  };

  // -------------------------- pack all weights ----------------------------
  for (int l = 0; l < 3; l++) {
    pack(cq_q[l].W, 64, pk_cqkvs[l], 0);
    pack(cq_k[l].W, 64, pk_cqkvs[l], 4);
    pack(cq_v[l].W, 64, pk_cqkvs[l], 8);
    pack(cq_s[l].W, 64, pk_cqkvs[l], 12);
    pack(cq_e[l].W, 64, pk_ce[l], 0);
  }
  for (int l = 0; l < 2; l++) pack(emlp[l].W, 192, pk_mlp[l], 0);
  pack(cq_lin_out.W, 64, pk_cout, 0);
  for (int l = 0; l < 6; l++) {
    pack(dg_q[l].W, 64, pk_dqkvs[l], 0);
    pack(dg_k[l].W, 64, pk_dqkvs[l], 4);
    pack(dg_v[l].W, 64, pk_dqkvs[l], 8);
    pack(dg_s[l].W, 64, pk_dqkvs[l], 12);
  }
  pack(dg_lin_out.W, 64, pk_dout, 0);

  // ============================ Clique branch =============================
  float* xn = qkvs;  // normalized node input staging
  float* zn = zb1;   // normalized edge input staging
  zero(stats, G * 12 * 2 + G);
  seg_stats_kernel<6><<<cdiv(N2, 256), 256, 0, stream>>>(cq_x, cq_batch, nullptr, N2, st_sum, st_sq, st_cnt);
  norm_apply_kernel<6><<<cdiv(N2 * 6, 256), 256, 0, stream>>>(cq_x, cq_batch, nullptr, N2, st_sum, st_sq, st_cnt, gnx_w, gnx_b, gnx_ms, xn);
  lin_small_kernel<<<cdiv(N2 * 64, 256), 256, 0, stream>>>(xn, N2, 6, cq_lin_in.W, cq_lin_in.b, xb);

  zero(stats, G * 12 * 2 + G);
  seg_stats_kernel<12><<<cdiv(E2, 256), 256, 0, stream>>>(cq_ea, cq_batch, erow, E2, st_sum, st_sq, st_cnt);
  norm_apply_kernel<12><<<cdiv(E2 * 12, 256), 256, 0, stream>>>(cq_ea, cq_batch, erow, E2, st_sum, st_sq, st_cnt, gne_w, gne_b, gne_ms, zn);
  lin_small_kernel<<<cdiv(E2 * 64, 256), 256, 0, stream>>>(zn, E2, 12, cq_lin_edge.W, cq_lin_edge.b, zb0);

  float* zcur = zb0;
  float* znext = zb1;
  for (int l = 0; l < 3; l++) {
    gemm_qkvs_wmma_kernel<<<cdiv(N2, 128), 256, 0, stream>>>(
        xb, pk_cqkvs[l], cq_q[l].b, cq_k[l].b, cq_v[l].b, cq_s[l].b, qkvs, N2);
    float* ep = znext;  // dead storage until edge-MLP rewrites it
    gemm64_wmma_kernel<false><<<cdiv(E2, 128), 256, 0, stream>>>(zcur, nullptr, pk_ce[l], cq_e[l].b, ep, E2);
    zero(umax, N2); zero(sums, N2); zero(aggr, (size_t)N2 * 64);
    attn_logits_kernel<<<cdiv(E2, 8), 256, 0, stream>>>(erow, ecol, qkvs, ep, logits, umax, E2);
    attn_weight_kernel<<<cdiv(E2, 256), 256, 0, stream>>>(ecol, logits, umax, sums, E2);
    attn_scatter_kernel<<<(int)(((long long)E2 * 64 + 255) / 256), 256, 0, stream>>>(erow, ecol, qkvs, ep, logits, aggr, E2);
    finalize_kernel<false><<<(int)(((long long)N2 * 64 + 255) / 256), 256, 0, stream>>>(aggr, sums, qkvs, nullptr, xb, N2);
    if (l < 2) {
      edge_mlp_wmma_kernel<<<cdiv(E2, 128), 256, 0, stream>>>(xb, zcur, erow, ecol, pk_mlp[l], emlp[l].b, znext, E2);
      float* t = zcur; zcur = znext; znext = t;
    }
  }
  gemm64_wmma_kernel<true><<<cdiv(P, 128), 256, 0, stream>>>(xb, cq_pick, pk_cout, cq_lin_out.b, xcq, P);

  // ============================== DAG branch ==============================
  zero(stats, G * 12 * 2 + G);
  seg_stats_kernel<10><<<cdiv(N1, 256), 256, 0, stream>>>(dag_x, dag_batch, nullptr, N1, st_sum, st_sq, st_cnt);
  norm_apply_kernel<10><<<cdiv(N1 * 10, 256), 256, 0, stream>>>(dag_x, dag_batch, nullptr, N1, st_sum, st_sq, st_cnt, nullptr, nullptr, nullptr, xn);
  lin_small_kernel<<<cdiv(N1 * 64, 256), 256, 0, stream>>>(xn, N1, 10, dg_lin_in.W, dg_lin_in.b, xb);

  for (int l = 0; l < 6; l++) {
    const int* ei = (l % 2 == 0) ? dag_eu : dag_ed;
    const unsigned char* msk = (l % 2 == 0) ? mask_up : mask_dn;
    const int* drow = ei;
    const int* dcol = ei + E1;
    gemm_qkvs_wmma_kernel<<<cdiv(N1, 128), 256, 0, stream>>>(
        xb, pk_dqkvs[l], dg_q[l].b, dg_k[l].b, dg_v[l].b, dg_s[l].b, qkvs, N1);
    zero(umax, N1); zero(sums, N1); zero(aggr, (size_t)N1 * 64);
    attn_logits_kernel<<<cdiv(E1, 8), 256, 0, stream>>>(drow, dcol, qkvs, nullptr, logits, umax, E1);
    attn_weight_kernel<<<cdiv(E1, 256), 256, 0, stream>>>(dcol, logits, umax, sums, E1);
    attn_scatter_kernel<<<(int)(((long long)E1 * 64 + 255) / 256), 256, 0, stream>>>(drow, dcol, qkvs, nullptr, logits, aggr, E1);
    finalize_kernel<true><<<(int)(((long long)N1 * 64 + 255) / 256), 256, 0, stream>>>(aggr, sums, qkvs, msk, xb, N1);
  }
  gemm64_wmma_kernel<true><<<cdiv(P, 128), 256, 0, stream>>>(xb, dag_pick, pk_dout, dg_lin_out.b, xdag, P);

  // ================================ tail ==================================
  tail_kernel<<<cdiv(P, 256), 256, 0, stream>>>(xcq, xdag, top.W, top.b, (float*)d_out, P);

  (void)n_in; (void)out_size; (void)ws_size;
}
